// MultiheadAttention_RoPE_63857573757037
// MI455X (gfx1250) — matrix operand order repaired and checked by reading, not compiled, not run on hardware
//
#include <hip/hip_runtime.h>

typedef __bf16 bf16;
typedef __attribute__((ext_vector_type(16))) __bf16 v16bf;
typedef __attribute__((ext_vector_type(8)))  float  v8f;
typedef __attribute__((ext_vector_type(4)))  unsigned int u32x4;

#define SEQ     2048
#define BATCH   2
#define NHEADS  16
#define HDIM    64
#define HIDDEN  1024
#define NTOK    (BATCH * SEQ)   // 4096 rows

// ---------------------------------------------------------------------------
// Cubic B-spline bases on the fixed uniform grid: g(i) = 0.4*i - 2.2, i=0..11
// ---------------------------------------------------------------------------
static __device__ __forceinline__ float gridpt(int i) { return 0.4f * (float)i - 2.2f; }

static __device__ __forceinline__ void bspline8(float x, float* out) {
  float b[11];
#pragma unroll
  for (int j = 0; j < 11; ++j)
    b[j] = (x >= gridpt(j) && x < gridpt(j + 1)) ? 1.0f : 0.0f;
#pragma unroll
  for (int k = 1; k <= 3; ++k) {
    float inv = 1.0f / (0.4f * (float)k);
#pragma unroll
    for (int j = 0; j < 10; ++j) {
      if (j < 11 - k) {
        float left  = (x - gridpt(j)) * inv;
        float right = (gridpt(j + k + 1) - x) * inv;
        b[j] = left * b[j] + right * b[j + 1];
      }
    }
  }
#pragma unroll
  for (int c = 0; c < 8; ++c) out[c] = b[c];
}

// ---------------------------------------------------------------------------
// WMMA fragment loaders (wave32, 16-bit operands, per ISA 7.12.2 layouts).
// A 16x32: lane m=l&15; lane<16 holds K[0..7],K[16..23]; lane>=16 K[8..15],K[24..31]
// B 32x16: lane n=l&15; lane<16 holds K[0..7],K[16..23]; lane>=16 K[8..15],K[24..31]
// Both operands row-major with K contiguous (stride ldk, multiple of 8).
// ---------------------------------------------------------------------------
static __device__ __forceinline__ v16bf load_frag_a(const bf16* __restrict__ base,
                                                    int ldk, int lane) {
  const int row = lane & 15, hi = lane >> 4;
  const bf16* p = base + (size_t)row * ldk + hi * 8;
  union { u32x4 u[2]; v16bf v; } r;
  r.u[0] = *(const u32x4*)(p);
  r.u[1] = *(const u32x4*)(p + 16);
  return r.v;
}

static __device__ __forceinline__ v16bf load_frag_b(const bf16* __restrict__ base,
                                                    int ldk, int lane) {
  const int col = lane & 15, hi = lane >> 4;
  const bf16* p = base + (size_t)col * ldk + hi * 8;
  union { u32x4 u[2]; v16bf v; } r;
  r.u[0] = *(const u32x4*)(p);
  r.u[1] = *(const u32x4*)(p + 16);
  return r.v;
}

static __device__ __forceinline__ v8f wmma_bf16(v16bf a, v16bf b, v8f c) {
  return __builtin_amdgcn_wmma_f32_16x16x32_bf16(false, a, false, b, (short)0, c,
                                                 false, false);
}

// Async global->LDS 16-byte copy (GLOBAL_LOAD_ASYNC_TO_LDS_B128, ASYNCcnt).
// lds operand is the wave-relative LDS byte address (low 32 bits of flat addr).
static __device__ __forceinline__ void async_copy16(void* lds, const void* gptr) {
  asm volatile("global_load_async_to_lds_b128 %0, %1, off"
               :: "v"((unsigned)(uintptr_t)lds), "v"(gptr) : "memory");
}

// ---------------------------------------------------------------------------
// KAN activation expansion: Act[n] = [ silu(x[n,:]) | bases(x[n,i])[0..7] ... ]
// ---------------------------------------------------------------------------
__global__ void kan_act_expand(const float* __restrict__ x0,
                               const float* __restrict__ x1,
                               const float* __restrict__ x2,
                               int seg, int Kin, bf16* __restrict__ Act) {
  int t = blockIdx.x * blockDim.x + threadIdx.x;
  if (t >= NTOK * Kin) return;
  int i = t % Kin;
  int n = t / Kin;
  int s = i / seg, loc = i - s * seg;
  const float* src = (s == 0) ? x0 : (s == 1) ? x1 : x2;
  float x = src[(size_t)n * seg + loc];
  size_t Kcat = (size_t)Kin * 9;
  bf16* row = Act + (size_t)n * Kcat;
  row[i] = (bf16)(x / (1.0f + __expf(-x)));      // silu
  float bs[8];
  bspline8(x, bs);
  bf16* sp = row + Kin + (size_t)i * 8;
#pragma unroll
  for (int c = 0; c < 8; ++c) sp[c] = (bf16)bs[c];
}

// Wcat[m] = [ base_w[m,:] | spline_w[m,i,c]*scaler[m,i] ... ] in bf16
__global__ void kan_weight_cat(const float* __restrict__ basew,
                               const float* __restrict__ splinew,
                               const float* __restrict__ scaler,
                               int M, int Kin, bf16* __restrict__ Wcat) {
  int t = blockIdx.x * blockDim.x + threadIdx.x;
  if (t >= M * Kin) return;
  int i = t % Kin, m = t / Kin;
  size_t Kcat = (size_t)Kin * 9;
  bf16* row = Wcat + (size_t)m * Kcat;
  size_t mi = (size_t)m * Kin + i;
  row[i] = (bf16)basew[mi];
  float sc = scaler[mi];
  const float* sw = splinew + mi * 8;
  bf16* dst = row + Kin + (size_t)i * 8;
#pragma unroll
  for (int c = 0; c < 8; ++c) dst[c] = (bf16)(sw[c] * sc);
}

// ---------------------------------------------------------------------------
// NT GEMM: C[N x M] = A[N x K] * W[M x K]^T, bf16 in / f32 out.
// Double-buffered async global->LDS staging (ASYNCcnt), 128x64 block tile,
// K-stage 64. 8 waves, each owns a 32x32 C tile (8 WMMAs per stage from LDS).
// N%128==0, M%64==0, K%64==0 (true for all call sites).
// ---------------------------------------------------------------------------
#define LDA 72   // 64 + 8 bf16 pad: keeps 16B alignment, spreads LDS banks
__global__ __launch_bounds__(256)
void gemm_bf16_nt(const bf16* __restrict__ A, const bf16* __restrict__ W,
                  float* __restrict__ C, int N, int M, int K) {
  __shared__ bf16 As[2][128 * LDA];
  __shared__ bf16 Ws[2][64 * LDA];

  const int tid  = threadIdx.x;
  const int lane = tid & 31;
  const int wid  = tid >> 5;
  const int wm = wid & 3, wn = wid >> 2;
  const size_t rowB = (size_t)blockIdx.x * 128;   // block row of A / C
  const size_t colB = (size_t)blockIdx.y * 64;    // block col of C = row of W

  // stage one 64-deep K slab into LDS buffer `buf` (6 async b128 per thread)
  auto stage = [&](int buf, int k0) {
    const bf16* Ag = A + rowB * K + k0;
    const bf16* Wg = W + colB * K + k0;
#pragma unroll
    for (int c = 0; c < 4; ++c) {              // A: 128 rows x 8 chunks
      int idx = tid + c * 256;
      int row = idx >> 3, ch = idx & 7;
      async_copy16(&As[buf][row * LDA + ch * 8], Ag + (size_t)row * K + ch * 8);
    }
#pragma unroll
    for (int c = 0; c < 2; ++c) {              // W: 64 rows x 8 chunks
      int idx = tid + c * 256;
      int row = idx >> 3, ch = idx & 7;
      async_copy16(&Ws[buf][row * LDA + ch * 8], Wg + (size_t)row * K + ch * 8);
    }
    __builtin_prefetch(Ag + 128, 0, 1);        // hint the slab after next
    __builtin_prefetch(Wg + 128, 0, 1);
  };

  v8f acc00 = {}, acc01 = {}, acc10 = {}, acc11 = {};
  const int nkt = K >> 6;

  stage(0, 0);
  for (int kt = 0; kt < nkt; ++kt) {
    const int cur = kt & 1;
    if (kt + 1 < nkt) {
      stage(cur ^ 1, (kt + 1) << 6);
      // 6 new loads in flight; <=6 outstanding means previous stage landed
      asm volatile("s_wait_asynccnt 0x6" ::: "memory");
    } else {
      asm volatile("s_wait_asynccnt 0x0" ::: "memory");
    }
    __syncthreads();

    const bf16* Abuf = &As[cur][0];
    const bf16* Wbuf = &Ws[cur][0];
#pragma unroll
    for (int kc = 0; kc < 64; kc += 32) {
      v16bf a0 = load_frag_a(Abuf + (wm * 32 + 0)  * LDA + kc, LDA, lane);
      v16bf a1 = load_frag_a(Abuf + (wm * 32 + 16) * LDA + kc, LDA, lane);
      v16bf b0 = load_frag_b(Wbuf + (wn * 32 + 0)  * LDA + kc, LDA, lane);
      v16bf b1 = load_frag_b(Wbuf + (wn * 32 + 16) * LDA + kc, LDA, lane);
      acc00 = wmma_bf16(a0, b0, acc00);
      acc01 = wmma_bf16(a0, b1, acc01);
      acc10 = wmma_bf16(a1, b0, acc10);
      acc11 = wmma_bf16(a1, b1, acc11);
    }
    __syncthreads();   // all waves done reading before next stage overwrites
  }

  // C/D layout: lane n = l&15; lanes>=16 hold rows 8..15; VGPR r -> row r(+8)
  const int cc = lane & 15, hi = lane >> 4;
  const size_t row0 = rowB + wm * 32, col0 = colB + wn * 32;
#pragma unroll
  for (int r = 0; r < 8; ++r) {
    int mrow = r + 8 * hi;
    C[(row0 + mrow) * M + col0 + cc]           = acc00[r];
    C[(row0 + mrow) * M + col0 + 16 + cc]      = acc01[r];
    C[(row0 + 16 + mrow) * M + col0 + cc]      = acc10[r];
    C[(row0 + 16 + mrow) * M + col0 + 16 + cc] = acc11[r];
  }
}

// ---------------------------------------------------------------------------
// RoPE + bf16 split: QKV (NTOK x 3072, per-head [q64|k64|v64]) ->
//   Qh,Kh [B,H,S,64] bf16 (rope applied), Vt [B,H,64,S] bf16 (transposed so
//   the P@V B-operand is K-contiguous too).
// ---------------------------------------------------------------------------
__global__ void rope_split(const float* __restrict__ QKV,
                           bf16* __restrict__ Qh, bf16* __restrict__ Kh,
                           bf16* __restrict__ Vt) {
  int t = blockIdx.x * blockDim.x + threadIdx.x;   // B*H*S*32 pairs
  if (t >= BATCH * NHEADS * SEQ * 32) return;
  int d2 = t & 31;
  int s  = (t >> 5) & (SEQ - 1);
  int bh = t >> 16;                 // SEQ*32 = 65536
  int b = bh >> 4, h = bh & 15;
  size_t n = (size_t)b * SEQ + s;
  const float* p = QKV + n * 3072 + h * 192;
  float q0 = p[2 * d2],       q1 = p[2 * d2 + 1];
  float k0 = p[64 + 2 * d2],  k1 = p[64 + 2 * d2 + 1];
  float v0 = p[128 + 2 * d2], v1 = p[128 + 2 * d2 + 1];
  float fr  = __powf(10000.0f, -(float)(2 * d2) * (1.0f / 64.0f));
  float ang = (float)s * fr;
  float cs = cosf(ang), sn = sinf(ang);
  size_t qk = ((size_t)bh * SEQ + s) * 64;
  Qh[qk + 2 * d2]     = (bf16)(q0 * cs - q1 * sn);
  Qh[qk + 2 * d2 + 1] = (bf16)(q1 * cs + q0 * sn);
  Kh[qk + 2 * d2]     = (bf16)(k0 * cs - k1 * sn);
  Kh[qk + 2 * d2 + 1] = (bf16)(k1 * cs + k0 * sn);
  size_t vb = (size_t)bh * 64;
  Vt[(vb + 2 * d2) * SEQ + s]     = (bf16)v0;
  Vt[(vb + 2 * d2 + 1) * SEQ + s] = (bf16)v1;
}

// ---------------------------------------------------------------------------
// Flash attention: per wave one 16-query tile, online softmax over 32-key
// tiles. All operands WMMA; P bounced through LDS (C-layout -> A-layout).
// grid = (B*H, SEQ/128), block = 256 (8 waves).
// ---------------------------------------------------------------------------
__global__ __launch_bounds__(256)
void flash_attn(const bf16* __restrict__ Qh, const bf16* __restrict__ Kh,
                const bf16* __restrict__ Vt, const float* __restrict__ mask,
                float* __restrict__ ctx) {
  __shared__ bf16 pbuf[8][16 * 32];
  const int lane = threadIdx.x & 31;
  const int wid  = threadIdx.x >> 5;
  const int bh = blockIdx.x;
  const int b = bh >> 4, h = bh & 15;
  const int q0 = (blockIdx.y * 8 + wid) * 16;
  const bf16* Qb = Qh + ((size_t)bh * SEQ + q0) * 64;
  const bf16* Kb = Kh + (size_t)bh * SEQ * 64;
  const bf16* Vb = Vt + (size_t)bh * 64 * SEQ;
  const int cc = lane & 15, hi = lane >> 4;

  v16bf aq0 = load_frag_a(Qb, 64, lane);       // Q d=0..31
  v16bf aq1 = load_frag_a(Qb + 32, 64, lane);  // Q d=32..63

  float rowm[8], rowl[8];
#pragma unroll
  for (int r = 0; r < 8; ++r) { rowm[r] = -1e30f; rowl[r] = 0.0f; }
  v8f o0 = {}, o1 = {}, o2 = {}, o3 = {};
  const float scale = 0.125f;  // 1/sqrt(64)

  for (int j = 0; j < SEQ; j += 32) {
    v8f slo = {}, shi = {};
    slo = wmma_bf16(aq0, load_frag_b(Kb + (size_t)j * 64,        64, lane), slo);
    slo = wmma_bf16(aq1, load_frag_b(Kb + (size_t)j * 64 + 32,   64, lane), slo);
    shi = wmma_bf16(aq0, load_frag_b(Kb + (size_t)(j + 16) * 64,      64, lane), shi);
    shi = wmma_bf16(aq1, load_frag_b(Kb + (size_t)(j + 16) * 64 + 32, 64, lane), shi);

#pragma unroll
    for (int r = 0; r < 8; ++r) {
      int m = r + 8 * hi;
      const float* mrow = mask + (size_t)(q0 + m) * SEQ + j;
      float vlo = slo[r] * scale + mrow[cc];
      float vhi = shi[r] * scale + mrow[16 + cc];
      float tmax = fmaxf(vlo, vhi);
      tmax = fmaxf(tmax, __shfl_xor(tmax, 1));
      tmax = fmaxf(tmax, __shfl_xor(tmax, 2));
      tmax = fmaxf(tmax, __shfl_xor(tmax, 4));
      tmax = fmaxf(tmax, __shfl_xor(tmax, 8));
      float nm   = fmaxf(rowm[r], tmax);
      float corr = __expf(rowm[r] - nm);
      float plo  = __expf(vlo - nm);
      float phi  = __expf(vhi - nm);
      float ps = plo + phi;
      ps += __shfl_xor(ps, 1);
      ps += __shfl_xor(ps, 2);
      ps += __shfl_xor(ps, 4);
      ps += __shfl_xor(ps, 8);
      rowl[r] = rowl[r] * corr + ps;
      rowm[r] = nm;
      o0[r] *= corr; o1[r] *= corr; o2[r] *= corr; o3[r] *= corr;
      pbuf[wid][m * 32 + cc]      = (bf16)plo;   // row-major 16x32 P tile
      pbuf[wid][m * 32 + 16 + cc] = (bf16)phi;
    }
    // intra-wave DS ops are in-order; just drain DScnt before reload
    asm volatile("s_wait_dscnt 0" ::: "memory");
    v16bf pf = load_frag_a(&pbuf[wid][0], 32, lane);
    o0 = wmma_bf16(pf, load_frag_b(Vb + (size_t)0  * SEQ + j, SEQ, lane), o0);
    o1 = wmma_bf16(pf, load_frag_b(Vb + (size_t)16 * SEQ + j, SEQ, lane), o1);
    o2 = wmma_bf16(pf, load_frag_b(Vb + (size_t)32 * SEQ + j, SEQ, lane), o2);
    o3 = wmma_bf16(pf, load_frag_b(Vb + (size_t)48 * SEQ + j, SEQ, lane), o3);
  }

#pragma unroll
  for (int r = 0; r < 8; ++r) {
    int m = r + 8 * hi;
    float inv = 1.0f / rowl[r];
    float* crow = ctx + ((size_t)b * SEQ + q0 + m) * HIDDEN + h * 64;
    crow[cc]      = o0[r] * inv;
    crow[16 + cc] = o1[r] * inv;
    crow[32 + cc] = o2[r] * inv;
    crow[48 + cc] = o3[r] * inv;
  }
}

// ---------------------------------------------------------------------------
extern "C" void kernel_launch(void* const* d_in, const int* in_sizes, int n_in,
                              void* d_out, int out_size, void* d_ws, size_t ws_size,
                              hipStream_t stream) {
  (void)in_sizes; (void)n_in; (void)out_size; (void)ws_size;
  const float* q            = (const float*)d_in[0];
  const float* k            = (const float*)d_in[1];
  const float* v            = (const float*)d_in[2];
  const float* attn_mask    = (const float*)d_in[3];
  const float* qkv_base_w   = (const float*)d_in[4];
  const float* qkv_spline_w = (const float*)d_in[5];
  const float* qkv_scaler   = (const float*)d_in[6];
  const float* out_base_w   = (const float*)d_in[7];
  const float* out_spline_w = (const float*)d_in[8];
  const float* out_scaler   = (const float*)d_in[9];

  // workspace carving (~508 MB total)
  char* ws = (char*)d_ws;
  size_t off = 0;
  auto carve = [&](size_t bytes) -> void* {
    void* p = ws + off;
    off += (bytes + 255) & ~(size_t)255;
    return p;
  };
  bf16*  Act   = (bf16*)carve((size_t)NTOK * 27648 * sizeof(bf16)); // reused for out-layer
  bf16*  WcatQ = (bf16*)carve((size_t)3072 * 27648 * sizeof(bf16));
  float* QKV   = (float*)carve((size_t)NTOK * 3072 * sizeof(float));
  bf16*  Qh    = (bf16*)carve((size_t)BATCH * NHEADS * SEQ * 64 * sizeof(bf16));
  bf16*  Kh    = (bf16*)carve((size_t)BATCH * NHEADS * SEQ * 64 * sizeof(bf16));
  bf16*  Vt    = (bf16*)carve((size_t)BATCH * NHEADS * 64 * SEQ * sizeof(bf16));
  float* ctx   = (float*)carve((size_t)NTOK * HIDDEN * sizeof(float));
  bf16*  WcatO = (bf16*)carve((size_t)1024 * 9216 * sizeof(bf16));

  // 1) qkv KAN layer: expand activations + pack weights, one big NT GEMM
  {
    int total = NTOK * 3072;
    kan_act_expand<<<(total + 255) / 256, 256, 0, stream>>>(q, k, v, 1024, 3072, Act);
  }
  {
    int total = 3072 * 3072;
    kan_weight_cat<<<(total + 255) / 256, 256, 0, stream>>>(
        qkv_base_w, qkv_spline_w, qkv_scaler, 3072, 3072, WcatQ);
  }
  gemm_bf16_nt<<<dim3(NTOK / 128, 3072 / 64), 256, 0, stream>>>(
      Act, WcatQ, QKV, NTOK, 3072, 27648);

  // 2) RoPE + head split (+ V transpose) to bf16
  {
    int total = BATCH * NHEADS * SEQ * 32;
    rope_split<<<(total + 255) / 256, 256, 0, stream>>>(QKV, Qh, Kh, Vt);
  }

  // 3) flash attention -> context (B,S,HIDDEN) fp32
  flash_attn<<<dim3(BATCH * NHEADS, SEQ / 128), 256, 0, stream>>>(
      Qh, Kh, Vt, attn_mask, ctx);

  // 4) output KAN layer
  {
    int total = NTOK * 1024;
    kan_act_expand<<<(total + 255) / 256, 256, 0, stream>>>(ctx, ctx, ctx, 1024, 1024, Act);
  }
  {
    int total = 1024 * 1024;
    kan_weight_cat<<<(total + 255) / 256, 256, 0, stream>>>(
        out_base_w, out_spline_w, out_scaler, 1024, 1024, WcatO);
  }
  gemm_bf16_nt<<<dim3(NTOK / 128, 1024 / 64), 256, 0, stream>>>(
      Act, WcatO, (float*)d_out, NTOK, 1024, 9216);
}
